// Mom_DoubleSelfAttnBlock_78391743086620
// MI455X (gfx1250) — matrix operand order repaired and checked by reading, not compiled, not run on hardware
//
#include <hip/hip_runtime.h>
#include <hip/hip_bf16.h>
#include <cstdint>
#include <cstddef>

// ---------------- problem constants (from reference) ----------------
#define DD    1024
#define HH    4
#define DKk   64
#define DVv   128
#define HDK   256
#define HDV   512
#define MM    4
#define M5    5
#define CONVK 4
#define TQ    512
#define TKV   1024
#define TTOT  1536
#define II    2816
#define II2   5632
#define EPSF  1e-6f

typedef __bf16 bf16;
typedef __attribute__((ext_vector_type(16))) __bf16 bf16x16;
typedef __attribute__((ext_vector_type(8)))  float  f32x8;
typedef __attribute__((ext_vector_type(4)))  int    v4i;

// ---- CDNA5 async global->LDS path (guarded; falls back to plain copies) ----
#if defined(__has_builtin)
#if __has_builtin(__builtin_amdgcn_global_load_async_to_lds_b128)
#define HAVE_ASYNC_LDS 1
#endif
#endif

__device__ __forceinline__ void async_copy16(const bf16* g, bf16* l) {
#if defined(HAVE_ASYNC_LDS)
  __builtin_amdgcn_global_load_async_to_lds_b128(
      (__attribute__((address_space(1))) v4i*)(uintptr_t)g,
      (__attribute__((address_space(3))) v4i*)(uint32_t)(uintptr_t)l,
      0, 0);
#else
  *(uint4*)l = *(const uint4*)g;
#endif
}

__device__ __forceinline__ void async_wait_all() {
#if defined(HAVE_ASYNC_LDS)
#if __has_builtin(__builtin_amdgcn_s_wait_asynccnt)
  __builtin_amdgcn_s_wait_asynccnt(0);
#else
  asm volatile("s_wait_asynccnt 0" ::: "memory");
#endif
#endif
}

// ---------------- elementwise helpers ----------------
__global__ void k_f32_to_bf16(const float* __restrict__ x, bf16* __restrict__ y, int n) {
  int i = blockIdx.x * blockDim.x + threadIdx.x;
  if (i < n) y[i] = (bf16)x[i];
}

__global__ void k_add(const float* __restrict__ a, const float* __restrict__ b,
                      float* __restrict__ c, int n) {
  int i = blockIdx.x * blockDim.x + threadIdx.x;
  if (i < n) c[i] = a[i] + b[i];
}

// rmsnorm over a row of `ncols`, output bf16 for WMMA consumption
__global__ void k_rmsnorm_bf16(const float* __restrict__ x, const float* __restrict__ w,
                               bf16* __restrict__ y, int ncols) {
  int row = blockIdx.x;
  const float* xr = x + (size_t)row * ncols;
  bf16* yr = y + (size_t)row * ncols;
  float s = 0.f;
  for (int i = threadIdx.x; i < ncols; i += blockDim.x) { float v = xr[i]; s += v * v; }
  __shared__ float red[256];
  red[threadIdx.x] = s; __syncthreads();
  for (int st = blockDim.x >> 1; st > 0; st >>= 1) {
    if ((int)threadIdx.x < st) red[threadIdx.x] += red[threadIdx.x + st];
    __syncthreads();
  }
  float scale = rsqrtf(red[0] / (float)ncols + EPSF);
  for (int i = threadIdx.x; i < ncols; i += blockDim.x)
    yr[i] = (bf16)(xr[i] * scale * w[i]);
}

// ---------------- LDS-tiled, double-buffered WMMA bf16 GEMM ----------------
// C[Mr,N] = A[Mr,K] @ B[K,N]; block 256 threads = 8 waves (2x4 wave grid),
// output tile 128x128, K-step 32, 8 v_wmma per wave per K-step.
// A panel staged async (global_load_async_to_lds_b128) row-major As[128][32];
// B panel staged transposed Bt[128][32] (contiguous ds_load_b128 fragments).
// Double-buffered: tile k+1 staged while tile k computes.
// WMMA VGPR layouts (A/B use contiguous k, a valid order chosen here; A and B match):
//   A frag: lane&15 = M row; half 0 holds K 0..15, half 1 K 16..31
//   B frag: lane&15 = N col; half 0 holds K 0..15, half 1 K 16..31
//   C/D:    vgpr r -> row r / r+8 per half-wave, lane&15 = col
#define BMT 128
#define BNT 128
#define BKT 32

__device__ __forceinline__ void stage_A(const bf16* Ablk, bf16* As, int K, int kb, int tid) {
  // 128x32 bf16 = 512 x 16B chunks, 2 per thread (unconditional -> no exec masking)
  int r0 = tid >> 2,          ko0 = (tid & 3) << 3;
  int r1 = (tid + 256) >> 2,  ko1 = (tid & 3) << 3;
  async_copy16(Ablk + (size_t)r0 * K + kb + ko0, &As[r0 * BKT + ko0]);
  async_copy16(Ablk + (size_t)r1 * K + kb + ko1, &As[r1 * BKT + ko1]);
}

__device__ __forceinline__ void load_B(const bf16* Bblk, int N, int kb, int tid, uint4* u) {
  // chunk c: krow = c&31, nc = c>>5 (8 consecutive columns per chunk)
  int c0 = tid, c1 = tid + 256;
  u[0] = *(const uint4*)(Bblk + (size_t)(kb + (c0 & 31)) * N + ((c0 >> 5) << 3));
  u[1] = *(const uint4*)(Bblk + (size_t)(kb + (c1 & 31)) * N + ((c1 >> 5) << 3));
}

__device__ __forceinline__ void store_B(bf16* Bt, int tid, const uint4* u) {
  int c0 = tid, c1 = tid + 256;
  const bf16* e0 = (const bf16*)&u[0];
  const bf16* e1 = (const bf16*)&u[1];
#pragma unroll
  for (int j = 0; j < 8; j++)
    Bt[(((c0 >> 5) << 3) + j) * BKT + (c0 & 31)] = e0[j];
#pragma unroll
  for (int j = 0; j < 8; j++)
    Bt[(((c1 >> 5) << 3) + j) * BKT + (c1 & 31)] = e1[j];
}

__global__ __launch_bounds__(256) void k_wmma_gemm(
    const bf16* __restrict__ A, const bf16* __restrict__ B,
    float* __restrict__ C, int Mr, int N, int K) {
  __shared__ bf16 As[2][BMT * BKT];   // 2 x 8 KB
  __shared__ bf16 Bt[2][BNT * BKT];   // 2 x 8 KB, [n][k]
  int nb = N >> 7;
  int bm = blockIdx.x / nb;
  int bn = blockIdx.x - bm * nb;
  int tid = threadIdx.x;
  int wave = tid >> 5, lane = tid & 31;
  int wm = wave >> 2, wn = wave & 3;       // 2 x 4 wave grid
  int lr = lane & 15, half = lane >> 4;

  const bf16* Ablk = A + (size_t)(bm * BMT) * K;
  const bf16* Bblk = B + bn * BNT;

  f32x8 zero = {};
  f32x8 acc[4][2];
#pragma unroll
  for (int i = 0; i < 4; i++)
#pragma unroll
    for (int j = 0; j < 2; j++) acc[i][j] = zero;

  // prologue: stage tile 0
  uint4 ub[2];
  stage_A(Ablk, As[0], K, 0, tid);
  load_B(Bblk, N, 0, tid, ub);
  store_B(Bt[0], tid, ub);
  async_wait_all();
  __syncthreads();

  int buf = 0;
  for (int kb = 0; kb < K; kb += BKT) {
    int nkb = kb + BKT;
    bool more = nkb < K;                       // wave-uniform
    if (more) {
      stage_A(Ablk, As[buf ^ 1], K, nkb, tid); // async; latency hides under compute
      load_B(Bblk, N, nkb, tid, ub);           // global loads in flight during compute
    }

    // ---- compute tile `buf`: per wave, 2 B frags + 4 A frags -> 8 wmma
    const bf16* Asb = As[buf];
    const bf16* Btb = Bt[buf];
    bf16x16 bfr[2];
#pragma unroll
    for (int in = 0; in < 2; in++) {
      int n = wn * 32 + in * 16 + lr;
      const bf16* bp = &Btb[n * BKT + half * 16];
#pragma unroll
      for (int i = 0; i < 16; i++) bfr[in][i] = bp[i];
    }
#pragma unroll
    for (int im = 0; im < 4; im++) {
      int r = wm * 64 + im * 16 + lr;
      const bf16* ap = &Asb[r * BKT + half * 16];
      bf16x16 af;
#pragma unroll
      for (int i = 0; i < 8; i++) { af[i] = ap[i]; af[8 + i] = ap[8 + i]; }
#pragma unroll
      for (int in = 0; in < 2; in++)
        acc[im][in] = __builtin_amdgcn_wmma_f32_16x16x32_bf16(
            false, af, false, bfr[in], (short)0, acc[im][in], false, false);
    }

    if (more) store_B(Bt[buf ^ 1], tid, ub);   // other buffer; no conflict with readers
    async_wait_all();
    __syncthreads();
    buf ^= 1;
  }

  // ---- store C
#pragma unroll
  for (int im = 0; im < 4; im++) {
#pragma unroll
    for (int in = 0; in < 2; in++) {
      int row0 = bm * BMT + wm * 64 + im * 16 + half * 8;
      int col = bn * BNT + wn * 32 + in * 16 + lr;
      float* cb = C + (size_t)row0 * N + col;
#pragma unroll
      for (int r = 0; r < 8; r++) cb[(size_t)r * N] = acc[im][in][r];
    }
  }
}

// ---------------- small projections: logits (Wg), beta-raw (Wb), a-raw (Wa); N=4 each ----------------
__global__ void k_proj_small(const bf16* __restrict__ X, const float* __restrict__ Wg,
                             const float* __restrict__ Wb, const float* __restrict__ Wa,
                             float* __restrict__ rl, float* __restrict__ braw,
                             float* __restrict__ araw, int T) {
  int gw = (blockIdx.x * blockDim.x + threadIdx.x) >> 5;
  int lane = threadIdx.x & 31;
  int t = gw / 12;
  int c = gw - t * 12;
  if (t >= T) return;
  const float* W; float* out;
  int col = c & 3;
  if (c < 4)      { W = Wg; out = rl; }
  else if (c < 8) { W = Wb; out = braw; }
  else            { W = Wa; out = araw; }
  const bf16* xr = X + (size_t)t * DD;
  float s = 0.f;
  for (int k = lane; k < DD; k += 32) s += (float)xr[k] * W[k * 4 + col];
  for (int o = 16; o > 0; o >>= 1) s += __shfl_xor(s, o, 32);
  if (lane == 0) out[t * 4 + col] = s;
}

// ---------------- gating: softmax, top-2, weights/comb, g = -exp(A)*softplus, beta = sigmoid ----------------
__global__ void k_gating(const float* __restrict__ rl, const float* __restrict__ braw,
                         const float* __restrict__ araw, const float* __restrict__ A_log,
                         const float* __restrict__ dt_bias,
                         float* __restrict__ comb, float* __restrict__ gm,
                         float* __restrict__ bm, int T) {
  int t = blockIdx.x * blockDim.x + threadIdx.x;
  if (t >= T) return;
  float lg[4], p[4];
  float mx = -1e30f;
  for (int m = 0; m < 4; m++) { lg[m] = rl[t * 4 + m]; mx = fmaxf(mx, lg[m]); }
  float ssum = 0.f;
  for (int m = 0; m < 4; m++) { p[m] = expf(lg[m] - mx); ssum += p[m]; }
  for (int m = 0; m < 4; m++) p[m] /= ssum;
  int i0 = 0;
  for (int m = 1; m < 4; m++) if (p[m] > p[i0]) i0 = m;
  int i1 = -1;
  for (int m = 0; m < 4; m++) { if (m == i0) continue; if (i1 < 0 || p[m] > p[i1]) i1 = m; }
  float w0 = p[i0], w1 = p[i1], wsum = w0 + w1;
  w0 /= wsum; w1 /= wsum;
  float wts[4] = {0.f, 0.f, 0.f, 0.f};
  wts[i0] += w0; wts[i1] += w1;
  float gh[4], bh[4];
  for (int h = 0; h < 4; h++) {
    bh[h] = 1.f / (1.f + expf(-braw[t * 4 + h]));
    float z = araw[t * 4 + h] + dt_bias[h];
    float sp = (z > 20.f) ? z : log1pf(expf(z));
    gh[h] = -expf(A_log[h]) * sp;
  }
  for (int m = 0; m < 5; m++) {
    float actm = (m == 4) ? 1.f : ((wts[m] > 0.f) ? 1.f : 0.f);
    comb[t * 5 + m] = (m == 4) ? 1.f : wts[m];
    for (int h = 0; h < 4; h++) {
      gm[t * 20 + m * 4 + h] = gh[h] * actm;
      bm[t * 20 + m * 4 + h] = bh[h] * actm;
    }
  }
}

// ---------------- causal depthwise conv (K=4) + silu ----------------
__global__ void k_dwconv_silu(const float* __restrict__ x, const float* __restrict__ w,
                              float* __restrict__ y, int T, int C) {
  int i = blockIdx.x * blockDim.x + threadIdx.x;
  if (i >= T * C) return;
  int t = i / C, c = i - t * C;
  float s = 0.f;
#pragma unroll
  for (int j = 0; j < CONVK; j++) {
    int tt = t - (CONVK - 1) + j;
    if (tt >= 0) s += x[(size_t)tt * C + c] * w[j * C + c];
  }
  y[i] = s / (1.f + expf(-s));
}

// ---------------- l2norm over DK per (t,h), in place ----------------
__global__ void k_l2norm(float* __restrict__ x, int T) {
  int i = blockIdx.x * blockDim.x + threadIdx.x;  // over T*H
  if (i >= T * HH) return;
  float* p = x + (size_t)i * DKk;
  float s = 0.f;
  for (int j = 0; j < DKk; j++) s += p[j] * p[j];
  float r = rsqrtf(s + 1e-6f);
  for (int j = 0; j < DKk; j++) p[j] *= r;
}

// ---------------- gated delta-rule scan: one block per (m,h); S[64][128] in registers ----------------
// thread tid: v = tid>>1 (0..127), k-half = tid&1 -> 32 state elems each; pair combine via shfl_xor(1)
__global__ __launch_bounds__(256) void k_scan(
    const float* __restrict__ qc, const float* __restrict__ kc,
    const float* __restrict__ vc, const float* __restrict__ gm,
    const float* __restrict__ bm, float* __restrict__ ot, int T) {
  int m = blockIdx.x >> 2;
  int h = blockIdx.x & 3;
  int tid = threadIdx.x;
  int v = tid >> 1;
  int k0 = (tid & 1) * 32;
  float S[32];
#pragma unroll
  for (int i = 0; i < 32; i++) S[i] = 0.f;
  __shared__ float ksh[DKk], qsh[DKk], vsh[DVv];
  for (int t = 0; t < T; t++) {
    if (tid < DKk) {
      ksh[tid] = kc[(size_t)t * HDK + h * DKk + tid];
      qsh[tid] = qc[(size_t)t * HDK + h * DKk + tid];
    } else if (tid < DKk + DVv) {
      vsh[tid - DKk] = vc[(size_t)t * HDV + h * DVv + (tid - DKk)];
    }
    __syncthreads();
    float eg = expf(gm[t * 20 + m * 4 + h]);
    float bb = bm[t * 20 + m * 4 + h];
    float part = 0.f;
#pragma unroll
    for (int i = 0; i < 32; i++) { S[i] *= eg; part += ksh[k0 + i] * S[i]; }
    float kv = part + __shfl_xor(part, 1, 32);
    float delta = bb * (vsh[v] - kv);
    float op = 0.f;
#pragma unroll
    for (int i = 0; i < 32; i++) { S[i] += ksh[k0 + i] * delta; op += qsh[k0 + i] * S[i]; }
    float o = op + __shfl_xor(op, 1, 32);
    if ((tid & 1) == 0) ot[(size_t)t * (M5 * HH * DVv) + (m * 4 + h) * DVv + v] = o;
    __syncthreads();
  }
}

// ---------------- combine memories, rmsnorm(DV) * silu(gate), emit bf16 ----------------
__global__ __launch_bounds__(128) void k_combine(
    const float* __restrict__ ot, const float* __restrict__ comb,
    const float* __restrict__ gate, const float* __restrict__ onorm,
    bf16* __restrict__ obf, int T) {
  int th = blockIdx.x;
  int t = th >> 2, h = th & 3;
  int v = threadIdx.x;
  float o = 0.f;
#pragma unroll
  for (int m = 0; m < 5; m++)
    o += comb[t * 5 + m] * ot[(size_t)t * (M5 * HH * DVv) + (m * 4 + h) * DVv + v];
  __shared__ float red[128];
  red[v] = o * o; __syncthreads();
  for (int st = 64; st > 0; st >>= 1) {
    if (v < st) red[v] += red[v + st];
    __syncthreads();
  }
  float y = o * rsqrtf(red[0] / (float)DVv + EPSF) * onorm[v];
  float g = gate[(size_t)t * HDV + h * DVv + v];
  y *= g / (1.f + expf(-g));
  obf[(size_t)t * HDV + h * DVv + v] = (bf16)y;
}

// ---------------- swiglu: silu(g)*u -> bf16 ----------------
__global__ void k_swiglu(const float* __restrict__ gu, bf16* __restrict__ h, int T) {
  int i = blockIdx.x * blockDim.x + threadIdx.x;
  if (i >= T * II) return;
  int t = i / II, c = i - t * II;
  float g = gu[(size_t)t * II2 + c];
  float u = gu[(size_t)t * II2 + II + c];
  h[i] = (bf16)((g / (1.f + expf(-g))) * u);
}

// =====================================================================
// host side
// =====================================================================
struct AttnScratch {
  float *qg, *kg, *vg, *gateg, *qc, *kc, *vc;
  float *braw, *araw, *comb, *gm, *bm, *ot;
  bf16  *obf;
};

static void launch_gemm(const bf16* A, const bf16* B, float* C,
                        int Mr, int N, int K, hipStream_t st) {
  int blocks = (Mr >> 7) * (N >> 7);            // 128x128 tiles
  k_wmma_gemm<<<blocks, 256, 0, st>>>(A, B, C, Mr, N, K);
}

static void run_attn(const bf16* xn, int T,
                     const float* Wg, const float* Wb, const float* Wa,
                     const float* cq, const float* ck, const float* cv,
                     const float* A_log, const float* dt_bias, const float* onorm,
                     const bf16* Wq_bf, const bf16* Wk_bf, const bf16* Wv_bf,
                     const bf16* Wog_bf, const bf16* Wo_bf,
                     float* rl_out, float* attn_out, AttnScratch* s, hipStream_t st) {
  launch_gemm(xn, Wq_bf, s->qg, T, HDK, DD, st);
  launch_gemm(xn, Wk_bf, s->kg, T, HDK, DD, st);
  launch_gemm(xn, Wv_bf, s->vg, T, HDV, DD, st);
  launch_gemm(xn, Wog_bf, s->gateg, T, HDV, DD, st);
  int waves = T * 12;
  k_proj_small<<<(waves + 7) / 8, 256, 0, st>>>(xn, Wg, Wb, Wa, rl_out, s->braw, s->araw, T);
  k_gating<<<(T + 255) / 256, 256, 0, st>>>(rl_out, s->braw, s->araw, A_log, dt_bias,
                                            s->comb, s->gm, s->bm, T);
  k_dwconv_silu<<<(T * HDK + 255) / 256, 256, 0, st>>>(s->qg, cq, s->qc, T, HDK);
  k_dwconv_silu<<<(T * HDK + 255) / 256, 256, 0, st>>>(s->kg, ck, s->kc, T, HDK);
  k_dwconv_silu<<<(T * HDV + 255) / 256, 256, 0, st>>>(s->vg, cv, s->vc, T, HDV);
  k_l2norm<<<(T * HH + 255) / 256, 256, 0, st>>>(s->qc, T);
  k_l2norm<<<(T * HH + 255) / 256, 256, 0, st>>>(s->kc, T);
  k_scan<<<M5 * HH, 256, 0, st>>>(s->qc, s->kc, s->vc, s->gm, s->bm, s->ot, T);
  k_combine<<<T * HH, 128, 0, st>>>(s->ot, s->comb, s->gateg, onorm, s->obf, T);
  launch_gemm(s->obf, Wo_bf, attn_out, T, DD, HDV, st);
}

extern "C" void kernel_launch(void* const* d_in, const int* in_sizes, int n_in,
                              void* d_out, int out_size, void* d_ws, size_t ws_size,
                              hipStream_t stream) {
  (void)in_sizes; (void)n_in; (void)out_size; (void)ws_size;

  const float* query       = (const float*)d_in[0];
  const float* keyval      = (const float*)d_in[1];
  const float* attn_norm_w = (const float*)d_in[2];
  const float* mlp_norm_w  = (const float*)d_in[3];
  const float* qkv_norm_w  = (const float*)d_in[4];
  const float* ffn_norm_w  = (const float*)d_in[5];
  const float* mlp_Wgu     = (const float*)d_in[6];
  const float* mlp_Wd      = (const float*)d_in[7];
  const float* ffn_Wgu     = (const float*)d_in[8];
  const float* ffn_Wd      = (const float*)d_in[9];

  // a_ params at 10..23, b_ params at 24..37 (same order)
  const float *aWg = (const float*)d_in[10], *aWq = (const float*)d_in[11],
              *aWk = (const float*)d_in[12], *aWv = (const float*)d_in[13],
              *acq = (const float*)d_in[14], *ack = (const float*)d_in[15],
              *acv = (const float*)d_in[16], *aWb = (const float*)d_in[17],
              *aWa = (const float*)d_in[18], *aAl = (const float*)d_in[19],
              *adt = (const float*)d_in[20], *aWog = (const float*)d_in[21],
              *aon = (const float*)d_in[22], *aWo = (const float*)d_in[23];
  const float *bWg = (const float*)d_in[24], *bWq = (const float*)d_in[25],
              *bWk = (const float*)d_in[26], *bWv = (const float*)d_in[27],
              *bcq = (const float*)d_in[28], *bck = (const float*)d_in[29],
              *bcv = (const float*)d_in[30], *bWb = (const float*)d_in[31],
              *bWa = (const float*)d_in[32], *bAl = (const float*)d_in[33],
              *bdt = (const float*)d_in[34], *bWog = (const float*)d_in[35],
              *bon = (const float*)d_in[36], *bWo = (const float*)d_in[37];

  float* out   = (float*)d_out;                    // (512,1024)
  float* rl_q  = out + (size_t)TQ * DD;            // (512,4)
  float* rl_kv = rl_q + (size_t)TQ * MM;           // (1536,4)

  // ---- bump allocator over d_ws ----
  char* base = (char*)d_ws;
  size_t off = 0;
  auto alloc = [&](size_t bytes) -> char* {
    char* p = base + off;
    off = (off + bytes + 255) & ~(size_t)255;
    return p;
  };

  // bf16 weight copies
  bf16* mlp_Wgu_bf = (bf16*)alloc((size_t)DD * II2 * 2);
  bf16* mlp_Wd_bf  = (bf16*)alloc((size_t)II * DD * 2);
  bf16* ffn_Wgu_bf = (bf16*)alloc((size_t)DD * II2 * 2);
  bf16* ffn_Wd_bf  = (bf16*)alloc((size_t)II * DD * 2);
  bf16 *aWq_bf = (bf16*)alloc((size_t)DD * HDK * 2), *aWk_bf = (bf16*)alloc((size_t)DD * HDK * 2);
  bf16 *aWv_bf = (bf16*)alloc((size_t)DD * HDV * 2), *aWog_bf = (bf16*)alloc((size_t)DD * HDV * 2);
  bf16 *aWo_bf = (bf16*)alloc((size_t)HDV * DD * 2);
  bf16 *bWq_bf = (bf16*)alloc((size_t)DD * HDK * 2), *bWk_bf = (bf16*)alloc((size_t)DD * HDK * 2);
  bf16 *bWv_bf = (bf16*)alloc((size_t)DD * HDV * 2), *bWog_bf = (bf16*)alloc((size_t)DD * HDV * 2);
  bf16 *bWo_bf = (bf16*)alloc((size_t)HDV * DD * 2);

  // activations (sized for T=1536 where shared between both attentions)
  bf16*  xn_bf = (bf16*)alloc((size_t)TTOT * DD * 2);
  AttnScratch s;
  s.qg    = (float*)alloc((size_t)TTOT * HDK * 4);
  s.kg    = (float*)alloc((size_t)TTOT * HDK * 4);
  s.vg    = (float*)alloc((size_t)TTOT * HDV * 4);
  s.gateg = (float*)alloc((size_t)TTOT * HDV * 4);
  s.qc    = (float*)alloc((size_t)TTOT * HDK * 4);
  s.kc    = (float*)alloc((size_t)TTOT * HDK * 4);
  s.vc    = (float*)alloc((size_t)TTOT * HDV * 4);
  s.braw  = (float*)alloc((size_t)TTOT * 4 * 4);
  s.araw  = (float*)alloc((size_t)TTOT * 4 * 4);
  s.comb  = (float*)alloc((size_t)TTOT * 5 * 4);
  s.gm    = (float*)alloc((size_t)TTOT * 20 * 4);
  s.bm    = (float*)alloc((size_t)TTOT * 20 * 4);
  s.ot    = (float*)alloc((size_t)TTOT * M5 * HH * DVv * 4);
  s.obf   = (bf16*)alloc((size_t)TTOT * HDV * 2);
  float* attn_out = (float*)alloc((size_t)TTOT * DD * 4);
  float* r1       = (float*)alloc((size_t)TQ * DD * 4);
  float* qout     = (float*)alloc((size_t)TQ * DD * 4);
  float* qkv      = (float*)alloc((size_t)TTOT * DD * 4);
  float* res2     = (float*)alloc((size_t)TQ * DD * 4);
  float* gu       = (float*)alloc((size_t)TQ * II2 * 4);
  bf16*  hbf      = (bf16*)alloc((size_t)TQ * II * 2);
  float* mlp_out  = (float*)alloc((size_t)TQ * DD * 4);

  auto cvt = [&](const float* src, bf16* dst, size_t n) {
    k_f32_to_bf16<<<((int)n + 255) / 256, 256, 0, stream>>>(src, dst, (int)n);
  };
  cvt(mlp_Wgu, mlp_Wgu_bf, (size_t)DD * II2);
  cvt(mlp_Wd,  mlp_Wd_bf,  (size_t)II * DD);
  cvt(ffn_Wgu, ffn_Wgu_bf, (size_t)DD * II2);
  cvt(ffn_Wd,  ffn_Wd_bf,  (size_t)II * DD);
  cvt(aWq, aWq_bf, (size_t)DD * HDK);  cvt(aWk, aWk_bf, (size_t)DD * HDK);
  cvt(aWv, aWv_bf, (size_t)DD * HDV);  cvt(aWog, aWog_bf, (size_t)DD * HDV);
  cvt(aWo, aWo_bf, (size_t)HDV * DD);
  cvt(bWq, bWq_bf, (size_t)DD * HDK);  cvt(bWk, bWk_bf, (size_t)DD * HDK);
  cvt(bWv, bWv_bf, (size_t)DD * HDV);  cvt(bWog, bWog_bf, (size_t)DD * HDV);
  cvt(bWo, bWo_bf, (size_t)HDV * DD);

  // ---- stage 1: attention A on query ----
  k_rmsnorm_bf16<<<TQ, 256, 0, stream>>>(query, attn_norm_w, xn_bf, DD);
  run_attn(xn_bf, TQ, aWg, aWb, aWa, acq, ack, acv, aAl, adt, aon,
           aWq_bf, aWk_bf, aWv_bf, aWog_bf, aWo_bf, rl_q, attn_out, &s, stream);
  k_add<<<(TQ * DD + 255) / 256, 256, 0, stream>>>(attn_out, query, r1, TQ * DD);

  // ---- stage 2: mlp ----
  k_rmsnorm_bf16<<<TQ, 256, 0, stream>>>(r1, mlp_norm_w, xn_bf, DD);
  launch_gemm(xn_bf, mlp_Wgu_bf, gu, TQ, II2, DD, stream);
  k_swiglu<<<(TQ * II + 255) / 256, 256, 0, stream>>>(gu, hbf, TQ);
  launch_gemm(hbf, mlp_Wd_bf, mlp_out, TQ, DD, II, stream);
  k_add<<<(TQ * DD + 255) / 256, 256, 0, stream>>>(mlp_out, r1, qout, TQ * DD);

  // ---- stage 3: attention B on [keyval ; qout] ----
  (void)hipMemcpyAsync(qkv, keyval, (size_t)TKV * DD * 4, hipMemcpyDeviceToDevice, stream);
  (void)hipMemcpyAsync(qkv + (size_t)TKV * DD, qout, (size_t)TQ * DD * 4,
                       hipMemcpyDeviceToDevice, stream);
  k_rmsnorm_bf16<<<TTOT, 256, 0, stream>>>(qkv, qkv_norm_w, xn_bf, DD);
  run_attn(xn_bf, TTOT, bWg, bWb, bWa, bcq, bck, bcv, bAl, bdt, bon,
           bWq_bf, bWk_bf, bWv_bf, bWog_bf, bWo_bf, rl_kv, attn_out, &s, stream);
  // qa = last TQ rows of attn_out
  k_add<<<(TQ * DD + 255) / 256, 256, 0, stream>>>(attn_out + (size_t)TKV * DD, qout,
                                                   res2, TQ * DD);

  // ---- stage 4: ffn ----
  k_rmsnorm_bf16<<<TQ, 256, 0, stream>>>(res2, ffn_norm_w, xn_bf, DD);
  launch_gemm(xn_bf, ffn_Wgu_bf, gu, TQ, II2, DD, stream);
  k_swiglu<<<(TQ * II + 255) / 256, 256, 0, stream>>>(gu, hbf, TQ);
  launch_gemm(hbf, ffn_Wd_bf, mlp_out, TQ, DD, II, stream);
  k_add<<<(TQ * DD + 255) / 256, 256, 0, stream>>>(mlp_out, res2, out, TQ * DD);
}
